// CAPMemory_6279242187176
// MI455X (gfx1250) — matrix operand order repaired and checked by reading, not compiled, not run on hardware
//
#include <hip/hip_runtime.h>
#include <stdint.h>

typedef __attribute__((ext_vector_type(16))) __bf16 v16bf;
typedef __attribute__((ext_vector_type(8)))  float  v8f;

#define INV_T   14.285714285714286f   // 1 / 0.07
#define NCAM    8
#define HARD_K  50
#define LAMDA   0.5f

static __device__ __forceinline__ uint16_t f32_to_bf16_rne(float x) {
  uint32_t u = __float_as_uint(x);
  u += 0x7fffu + ((u >> 16) & 1u);
  return (uint16_t)(u >> 16);
}

// Pack two f32 -> two bf16 in one dword: bias-add round + v_perm_b32 byte pack.
static __device__ __forceinline__ uint32_t pack2_bf16(float lo, float hi) {
  const uint32_t ul = __float_as_uint(lo) + 0x8000u;
  const uint32_t uh = __float_as_uint(hi) + 0x8000u;
  // result = { uh[31:16], ul[31:16] } : selector bytes {7,6,3,2}, src0=uh, src1=ul
  return __builtin_amdgcn_perm(uh, ul, 0x07060302u);
}

// ---------------------------------------------------------------------------
// Kernel A: L2-normalize each feats row, store bf16 (feats are reused 32000x,
// so one-time conversion pays; centers are NOT pre-converted -> stream once).
// ---------------------------------------------------------------------------
__global__ __launch_bounds__(256)
void cap_normalize_bf16(const float* __restrict__ feats,
                        uint16_t* __restrict__ fbf, int d) {
  const int row = blockIdx.x;
  const int tid = threadIdx.x;
  const float* src = feats + (size_t)row * d;
  uint16_t*   dst = fbf   + (size_t)row * d;

  float ss = 0.f;
  for (int k = tid; k < d; k += 256) { float v = src[k]; ss = fmaf(v, v, ss); }
  __shared__ float red[256];
  red[tid] = ss;
  __syncthreads();
  for (int off = 128; off > 0; off >>= 1) {
    if (tid < off) red[tid] += red[tid + off];
    __syncthreads();
  }
  const float inv = rsqrtf(red[0]);
  for (int k = tid; k < d; k += 256) dst[k] = f32_to_bf16_rne(src[k] * inv);
}

// ---------------------------------------------------------------------------
// Kernel B: fused GEMM (V_WMMA_F32_16X16X32_BF16) + scale/exp/masks/reduce.
// Block tile: ALL 256 samples x 64 centers -> `centers` streams from HBM once.
// Wave tile: 128 rows x 16 cols = 8 accumulators; B fragment unique per wave.
// Templated on D so A-tile addressing folds into immediate offsets
// (rt*16*D*2 bytes <= 448KB < 24-bit IOFFSET) instead of 8 live u64 pointers.
// Assumes n == 256, D % 32 == 0, M % 64 == 0.
// ---------------------------------------------------------------------------
template <int DD>
__global__ __launch_bounds__(256)
void cap_sims_wmma(const uint16_t* __restrict__ fbf,     // [256][d] bf16 unit rows
                   const float*    __restrict__ centers, // [M][d] f32 unit rows
                   const int*      __restrict__ labels,
                   const int*      __restrict__ camids,
                   float* __restrict__ own_g,            // [256]
                   float* __restrict__ part_intra,       // [256][gx]
                   float* __restrict__ part_inter,       // [256][gx]
                   int dRuntime, int gx) {
  const int d = DD ? DD : dRuntime;     // compile-time when specialized
  const int bx   = blockIdx.x;          // 64-center tile
  const int tid  = threadIdx.x;
  const int wave = tid >> 5;
  const int lane = tid & 31;
  const int colTile = wave & 3;         // 4 x 16-center tiles
  const int rowHalf = wave >> 2;        // 2 x 128-sample halves
  const int halfHi  = lane >> 4;
  const int lane16  = lane & 15;

  __shared__ int   s_lab[256], s_cam[256], s_own[256];
  __shared__ float s_pi[4][256], s_pj[4][256];

  {
    const int lb = labels[tid];
    const int cm = camids[tid];
    s_lab[tid] = lb; s_cam[tid] = cm; s_own[tid] = lb * NCAM + cm;
  }
  __syncthreads();

  const int rBase = rowHalf * 128;
  const int cBase = bx * 64 + colTile * 16;

  // A layout (16x32 bf16): lane<16 -> K {0..7,16..23}; lane>=16 -> {8..15,24..31}
  const uint16_t* aBase = fbf + (size_t)(rBase + lane16) * d + halfHi * 8;
  const int aTile = 16 * d;             // elements between row tiles
  // B layout (32x16 bf16): lane<16 -> K {0..7,16..23} of column lane16; lane>=16 -> {8..15,24..31}
  const float* bPtr = centers + (size_t)(cBase + lane16) * d + halfHi * 8;

  union AB { uint4 q[2]; v16bf v; };
  v8f acc[8] = {};

  for (int kk = 0; kk < d; kk += 32) {
    const float* bp = bPtr + kk;
    // prefetch the B stream two K-steps ahead (speculative, line-granular)
    __builtin_prefetch(bp + 64, 0, 0);

    const float4 f0 = *(const float4*)(bp + 0);
    const float4 f1 = *(const float4*)(bp + 4);
    const float4 f2 = *(const float4*)(bp + 16);
    const float4 f3 = *(const float4*)(bp + 20);

    // batch 0 of A fragments: issue loads, overlap with B conversion
    AB A0[4], A1[4];
#pragma unroll
    for (int rt = 0; rt < 4; ++rt) {
      const uint16_t* ap = aBase + rt * aTile + kk;
      A0[rt].q[0] = *(const uint4*)(ap);
      A0[rt].q[1] = *(const uint4*)(ap + 16);
    }

    union { uint32_t u[8]; v16bf v; } B;
    B.u[0] = pack2_bf16(f0.x, f0.y);
    B.u[1] = pack2_bf16(f0.z, f0.w);
    B.u[2] = pack2_bf16(f1.x, f1.y);
    B.u[3] = pack2_bf16(f1.z, f1.w);
    B.u[4] = pack2_bf16(f2.x, f2.y);
    B.u[5] = pack2_bf16(f2.z, f2.w);
    B.u[6] = pack2_bf16(f3.x, f3.y);
    B.u[7] = pack2_bf16(f3.z, f3.w);

    // batch 1 of A fragments in flight while batch 0 WMMAs execute
#pragma unroll
    for (int rt = 0; rt < 4; ++rt) {
      const uint16_t* ap = aBase + (rt + 4) * aTile + kk;
      A1[rt].q[0] = *(const uint4*)(ap);
      A1[rt].q[1] = *(const uint4*)(ap + 16);
    }

#pragma unroll
    for (int rt = 0; rt < 4; ++rt)
      acc[rt] = __builtin_amdgcn_wmma_f32_16x16x32_bf16(
          false, A0[rt].v, false, B.v, (short)0, acc[rt], false, false);
#pragma unroll
    for (int rt = 0; rt < 4; ++rt)
      acc[rt + 4] = __builtin_amdgcn_wmma_f32_16x16x32_bf16(
          false, A1[rt].v, false, B.v, (short)0, acc[rt + 4], false, false);
  }

  // ---- epilogue: scale, exp, masks, deterministic reductions ----
  const int m = cBase + lane16;                    // this lane's center index
#pragma unroll
  for (int rt = 0; rt < 8; ++rt) {
    float pi[8], pj[8];
#pragma unroll
    for (int v = 0; v < 8; ++v) {
      const int rLocal = rBase + rt * 16 + v + 8 * halfHi;
      const float s = acc[rt][v] * INV_T;
      const float e = __expf(s);
      const int lb = s_lab[rLocal];
      const int cm = s_cam[rLocal];
      pi[v] = ((m & (NCAM - 1)) == cm) ? e : 0.f;          // intra-cam denom
      const bool isLbl = (m >> 3) == lb;                   // same-label center
      const int  ovl   = min(max(m + 1 - lb * NCAM, 0), NCAM);
      const bool hard  = (!isLbl) && ((m + 1 - ovl) <= HARD_K);
      pj[v] = (isLbl || hard) ? e : 0.f;                   // inter denom
      if (m == s_own[rLocal]) own_g[rLocal] = s;           // unique writer
    }
    // reduce across the 16 lanes of each half (columns of this tile)
#pragma unroll
    for (int v = 0; v < 8; ++v) {
      float a = pi[v], b = pj[v];
#pragma unroll
      for (int msk = 8; msk > 0; msk >>= 1) {
        a += __shfl_xor(a, msk, 32);
        b += __shfl_xor(b, msk, 32);
      }
      if (lane16 == 0) {
        const int rLocal = rBase + rt * 16 + v + 8 * halfHi;
        s_pi[colTile][rLocal] = a;
        s_pj[colTile][rLocal] = b;
      }
    }
  }
  __syncthreads();
  // fixed-order combine of the 4 col tiles, one thread per sample
  part_intra[(size_t)tid * gx + bx] =
      s_pi[0][tid] + s_pi[1][tid] + s_pi[2][tid] + s_pi[3][tid];
  part_inter[(size_t)tid * gx + bx] =
      s_pj[0][tid] + s_pj[1][tid] + s_pj[2][tid] + s_pj[3][tid];
}

// ---------------------------------------------------------------------------
// Kernel C: fixed-order reduction of partials, per-sample losses, segment
// means (mean over segment == sum_i loss_i / count(segment_i)), outputs.
// ---------------------------------------------------------------------------
__global__ void cap_finalize(const float* __restrict__ own_g,
                             const float* __restrict__ part_intra,
                             const float* __restrict__ part_inter,
                             const int* __restrict__ labels,
                             const int* __restrict__ camids,
                             float* __restrict__ out,
                             int n, int gx, int out_size) {
  __shared__ int   s_lab[256], s_cam[256];
  __shared__ float s_red[256];
  const int tid = threadIdx.x;
  s_lab[tid] = labels[tid];
  s_cam[tid] = camids[tid];
  __syncthreads();

  float di = 0.f, dj = 0.f;
  const float* prI = part_intra + (size_t)tid * gx;
  const float* prJ = part_inter + (size_t)tid * gx;
  for (int b = 0; b < gx; ++b) { di += prI[b]; dj += prJ[b]; }

  const float ow = own_g[tid];
  const float li = ow - __logf(di);
  const float lj = ow - __logf(dj);

  int cc = 0, cl = 0;
  const int myc = s_cam[tid], myl = s_lab[tid];
  for (int j = 0; j < n; ++j) { cc += (s_cam[j] == myc); cl += (s_lab[j] == myl); }

  s_red[tid] = li / (float)cc;
  __syncthreads();
  for (int off = 128; off > 0; off >>= 1) {
    if (tid < off) s_red[tid] += s_red[tid + off];
    __syncthreads();
  }
  const float sum_intra = s_red[0];
  __syncthreads();

  s_red[tid] = lj / (float)cl;
  __syncthreads();
  for (int off = 128; off > 0; off >>= 1) {
    if (tid < off) s_red[tid] += s_red[tid + off];
    __syncthreads();
  }
  const float sum_inter = s_red[0];

  if (tid == 0) {
    out[0] = -sum_intra;                            // loss_intra
    if (out_size > 1) out[1] = -LAMDA * sum_inter;  // lambda * loss_inter
  }
}

// ---------------------------------------------------------------------------
extern "C" void kernel_launch(void* const* d_in, const int* in_sizes, int n_in,
                              void* d_out, int out_size, void* d_ws, size_t ws_size,
                              hipStream_t stream) {
  const float* feats   = (const float*)d_in[0];
  const float* centers = (const float*)d_in[1];
  const int*   labels  = (const int*)d_in[2];
  const int*   camids  = (const int*)d_in[3];
  (void)n_in; (void)ws_size;

  const int n  = in_sizes[2];            // 256
  const int d  = in_sizes[0] / n;        // 2048
  const int M  = in_sizes[1] / d;        // 32000
  const int gx = M / 64;                 // 500 column tiles

  // workspace carve-up (all regions fully overwritten every call)
  char* ws = (char*)d_ws;
  uint16_t* fbf = (uint16_t*)ws;
  size_t off = ((size_t)n * d * sizeof(uint16_t) + 255) & ~(size_t)255;
  float* own_g = (float*)(ws + off);
  off += ((size_t)n * sizeof(float) + 255) & ~(size_t)255;
  float* part_intra = (float*)(ws + off);
  off += ((size_t)n * gx * sizeof(float) + 255) & ~(size_t)255;
  float* part_inter = (float*)(ws + off);

  cap_normalize_bf16<<<n, 256, 0, stream>>>(feats, fbf, d);

  if (d == 2048) {
    cap_sims_wmma<2048><<<dim3(gx), 256, 0, stream>>>(
        fbf, centers, labels, camids, own_g, part_intra, part_inter, d, gx);
  } else {
    cap_sims_wmma<0><<<dim3(gx), 256, 0, stream>>>(
        fbf, centers, labels, camids, own_g, part_intra, part_inter, d, gx);
  }

  cap_finalize<<<1, n, 0, stream>>>(own_g, part_intra, part_inter,
                                    labels, camids, (float*)d_out, n, gx, out_size);
}
